// MyLSTM_73323681677973
// MI455X (gfx1250) — matrix operand order repaired and checked by reading, not compiled, not run on hardware
//
#include <hip/hip_runtime.h>
#include <math.h>

// ---------------------------------------------------------------------------
// CDNA5 / gfx1250 LSTM: persistent-per-batch-tile recurrence with bf16 WMMA.
//   grid = 8 WGs (batch tiles of 16 rows), block = 512 threads = 16 wave32s.
//   Per step: A = [X_t | H] (16x768 bf16, LDS)  x  Wpack (768x2048 bf16, L2)
//   Gate math + C-state kept in registers; H written back to LDS as bf16.
// Workspace: 3 MB packed gate weights + 256 KB packed Whq (WMMA B-fragment
// order), written by prep kernels each call (idempotent, deterministic).
// ---------------------------------------------------------------------------

typedef __attribute__((ext_vector_type(16))) __bf16 v16bf;
typedef __attribute__((ext_vector_type(8)))  __bf16 v8bf;
typedef __attribute__((ext_vector_type(8)))  float  v8f;

#define LSTM_H   512
#define KTOT     768           // 256 input + 512 hidden
#define NKK      24            // KTOT / 32
#define A_STRIDE 776           // 768 + 8 bf16 pad -> 388 dwords, 388%64=4 (no bank conflicts)

__device__ __forceinline__ float fast_sigmoid(float x) {
    return 1.0f / (1.0f + __expf(-x));
}
__device__ __forceinline__ float fast_tanh(float x) {
    float xc = fminf(fmaxf(x, -15.0f), 15.0f);
    float e  = __expf(-2.0f * xc);
    return (1.0f - e) / (1.0f + e);
}

// ---------------------------------------------------------------------------
// Prep: pack [Wx;Wh] (K=768, N=2048 = 4 gates x 512) into per-lane B-fragment
// order: block(w,kk,t) of 1024B; lane holds 16 bf16:
//   n = tile_base + lane%16,  k = kk*32 + 16*(lane/16) + j   (dense 32x16 B)
// Layout: [(w*24 + kk)*8 + t] blocks of 32 lanes * 32B.
// ---------------------------------------------------------------------------
__global__ void lstm_pack_gates(const float* __restrict__ Wxi, const float* __restrict__ Whi,
                                const float* __restrict__ Wxf, const float* __restrict__ Whf,
                                const float* __restrict__ Wxo, const float* __restrict__ Who,
                                const float* __restrict__ Wxc, const float* __restrict__ Whc,
                                __bf16* __restrict__ wpack)
{
    int q = blockIdx.x * blockDim.x + threadIdx.x;   // 16*24*8*32 = 98304 threads
    if (q >= 16 * 24 * 8 * 32) return;
    int lane = q & 31;
    int t    = (q >> 5) & 7;          // tile within wave: t = gate*2 + subtile
    int kk   = (q >> 8) % 24;
    int w    = q / (256 * 24);        // wave id 0..15
    int g    = t >> 1;
    int h    = w * 32 + (t & 1) * 16 + (lane & 15);
    int khalf = lane >> 4;
    const float* Wx[4] = {Wxi, Wxf, Wxo, Wxc};
    const float* Wh[4] = {Whi, Whf, Who, Whc};
    __bf16* dst = wpack + (size_t)q * 16;
#pragma unroll
    for (int j = 0; j < 16; ++j) {
        int k = kk * 32 + khalf * 16 + j;
        float v = (k < 256) ? Wx[g][(size_t)k * LSTM_H + h]
                            : Wh[g][(size_t)(k - 256) * LSTM_H + h];
        dst[j] = (__bf16)v;
    }
}

// Pack Whq (512x256): blocks [(nt*16 + kk)] of 1024B, same lane rule.
__global__ void lstm_pack_whq(const float* __restrict__ Whq, __bf16* __restrict__ qpack)
{
    int q = blockIdx.x * blockDim.x + threadIdx.x;   // 16*16*32 = 8192 threads
    if (q >= 16 * 16 * 32) return;
    int lane = q & 31;
    int kk   = (q >> 5) & 15;
    int nt   = q >> 9;
    int n    = nt * 16 + (lane & 15);
    int khalf = lane >> 4;
    __bf16* dst = qpack + (size_t)q * 16;
#pragma unroll
    for (int j = 0; j < 16; ++j) {
        int k = kk * 32 + khalf * 16 + j;
        dst[j] = (__bf16)Whq[(size_t)k * 256 + n];
    }
}

// ---------------------------------------------------------------------------
// Persistent recurrent kernel.
// ---------------------------------------------------------------------------
__global__ __launch_bounds__(512, 1)
void lstm_persistent(const float* __restrict__ X,          // [128,1024,256] f32
                     const v16bf* __restrict__ wpack,      // packed gate weights
                     const v16bf* __restrict__ qpack,      // packed Whq
                     const float* __restrict__ bi, const float* __restrict__ bf_,
                     const float* __restrict__ bo, const float* __restrict__ bc,
                     const float* __restrict__ bq,
                     float* __restrict__ out)              // [128,256] f32
{
    __shared__ __bf16 Abuf[16][A_STRIDE];   // [X_t bf16 (cols 0..255) | H bf16 (cols 256..767)]

    const int tid   = threadIdx.x;
    const int lane  = tid & 31;
    const int w     = tid >> 5;      // wave 0..15; also the X row this thread loads
    const int lanen = lane & 15;     // N index within 16x16 tile / matrix row for A
    const int half  = lane >> 4;
    const int b0    = blockIdx.x * 16;

    const v8f zero8 = {0.f, 0.f, 0.f, 0.f, 0.f, 0.f, 0.f, 0.f};

    // Zero A buffer: H starts at 0 (and pad columns are benign).
    {
        __bf16* flat = &Abuf[0][0];
        for (int i = tid; i < 16 * A_STRIDE; i += 512) flat[i] = (__bf16)0.0f;
    }

    // Per-thread gate biases: tile t covers n = w*32 + (t&1)*16 + lanen of gate t>>1.
    float bias[8];
    {
        const float* bg[4] = {bi, bf_, bo, bc};
#pragma unroll
        for (int t = 0; t < 8; ++t)
            bias[t] = bg[t >> 1][w * 32 + (t & 1) * 16 + lanen];
    }

    // Cell state lives in registers for all 1024 steps (2 h-subtiles).
    v8f Cst[2];
    Cst[0] = zero8;
    Cst[1] = zero8;

    const int   c8   = lane * 8;                                  // X column chunk
    const float* xrow = X + ((size_t)(b0 + w) * 1024) * 256 + c8; // this thread's batch row

#pragma unroll 1
    for (int tstep = 0; tstep < 1024; ++tstep) {
        // ---- stage X_t tile into LDS as bf16 (16 rows x 256 cols) ----
        const float4* xp = (const float4*)(xrow + (size_t)tstep * 256);
        float4 x0 = xp[0];
        float4 x1 = xp[1];
        v8bf xb;
        xb[0] = (__bf16)x0.x; xb[1] = (__bf16)x0.y; xb[2] = (__bf16)x0.z; xb[3] = (__bf16)x0.w;
        xb[4] = (__bf16)x1.x; xb[5] = (__bf16)x1.y; xb[6] = (__bf16)x1.z; xb[7] = (__bf16)x1.w;
        *(v8bf*)&Abuf[w][c8] = xb;
        if (tstep + 1 < 1024)
            __builtin_prefetch(xrow + (size_t)(tstep + 1) * 256, 0, 0);   // global_prefetch_b8

        __syncthreads();   // X staged + previous step's H visible

        // ---- P = A(16x768) @ Wpack(768 x [this wave's 8 n-tiles]) ----
        v8f acc[8];
#pragma unroll
        for (int t = 0; t < 8; ++t) acc[t] = zero8;

#pragma unroll 1
        for (int kk = 0; kk < NKK; ++kk) {
            // A fragment: row = lanen; k = 16*half + s (contiguous, matches packed B).
            const int c1 = kk * 32 + half * 16;
            v8bf alo = *(const v8bf*)&Abuf[lanen][c1];
            v8bf ahi = *(const v8bf*)&Abuf[lanen][c1 + 8];
            v16bf afrag = __builtin_shufflevector(alo, ahi,
                          0,1,2,3,4,5,6,7,8,9,10,11,12,13,14,15);
            const v16bf* bp = wpack + (((size_t)w * NKK + kk) * 8) * 32 + lane;
#pragma unroll
            for (int t = 0; t < 8; ++t) {
                v16bf bfrag = bp[(size_t)t * 32];
                acc[t] = __builtin_amdgcn_wmma_f32_16x16x32_bf16(
                    false, afrag, false, bfrag, (short)0, acc[t], false, false);
            }
        }

        __syncthreads();   // all waves done reading Abuf before H overwrite

        // ---- gates + state update, all in registers; write H back as bf16 ----
#pragma unroll
        for (int s = 0; s < 2; ++s) {
            v8f pi = acc[0 + s], pf = acc[2 + s], po = acc[4 + s], pc = acc[6 + s];
            const int hcol = 256 + w * 32 + s * 16 + lanen;
#pragma unroll
            for (int e = 0; e < 8; ++e) {
                float vi = fast_sigmoid(pi[e] + bias[0 + s]);
                float vf = fast_sigmoid(pf[e] + bias[2 + s]);
                float vo = fast_sigmoid(po[e] + bias[4 + s]);
                float vc = fast_tanh   (pc[e] + bias[6 + s]);
                float cn = vf * Cst[s][e] + vi * vc;
                Cst[s][e] = cn;
                float hn = vo * fast_tanh(cn);
                Abuf[e + 8 * half][hcol] = (__bf16)hn;   // row m = e + 8*half
            }
        }
        // next iteration's first __syncthreads orders these H stores vs. A reads
    }

    __syncthreads();   // final H committed to LDS

    // ---- out = H_T @ Whq + bq ; wave w owns output columns [16w, 16w+16) ----
    v8f accq = zero8;
#pragma unroll 1
    for (int kk = 0; kk < 16; ++kk) {
        const int c1 = 256 + kk * 32 + half * 16;
        v8bf alo = *(const v8bf*)&Abuf[lanen][c1];
        v8bf ahi = *(const v8bf*)&Abuf[lanen][c1 + 8];
        v16bf afrag = __builtin_shufflevector(alo, ahi,
                      0,1,2,3,4,5,6,7,8,9,10,11,12,13,14,15);
        v16bf bfrag = qpack[((size_t)w * 16 + kk) * 32 + lane];
        accq = __builtin_amdgcn_wmma_f32_16x16x32_bf16(
            false, afrag, false, bfrag, (short)0, accq, false, false);
    }
    const float bqv = bq[w * 16 + lanen];
#pragma unroll
    for (int e = 0; e < 8; ++e) {
        out[(size_t)(b0 + e + 8 * half) * 256 + w * 16 + lanen] = accq[e] + bqv;
    }
}

// ---------------------------------------------------------------------------
extern "C" void kernel_launch(void* const* d_in, const int* in_sizes, int n_in,
                              void* d_out, int out_size, void* d_ws, size_t ws_size,
                              hipStream_t stream)
{
    const float* X   = (const float*)d_in[0];
    const float* Wxi = (const float*)d_in[1];
    const float* Whi = (const float*)d_in[2];
    const float* Wxf = (const float*)d_in[3];
    const float* Whf = (const float*)d_in[4];
    const float* Wxo = (const float*)d_in[5];
    const float* Who = (const float*)d_in[6];
    const float* Wxc = (const float*)d_in[7];
    const float* Whc = (const float*)d_in[8];
    const float* bi  = (const float*)d_in[9];
    const float* bf  = (const float*)d_in[10];
    const float* bo  = (const float*)d_in[11];
    const float* bc  = (const float*)d_in[12];
    const float* Whq = (const float*)d_in[13];
    const float* bq  = (const float*)d_in[14];

    __bf16* wpack = (__bf16*)d_ws;                                   // 3 MB
    __bf16* qpack = (__bf16*)((char*)d_ws + (size_t)3 * 1024 * 1024); // 256 KB

    lstm_pack_gates<<<384, 256, 0, stream>>>(Wxi, Whi, Wxf, Whf, Wxo, Who, Wxc, Whc, wpack);
    lstm_pack_whq<<<32, 256, 0, stream>>>(Whq, qpack);
    lstm_persistent<<<8, 512, 0, stream>>>(X, (const v16bf*)wpack, (const v16bf*)qpack,
                                           bi, bf, bo, bc, bq, (float*)d_out);
}
